// Inference_and_Generation_16277926051842
// MI455X (gfx1250) — matrix operand order repaired and checked by reading, not compiled, not run on hardware
//
#include <hip/hip_runtime.h>
#include <math.h>

// ---------------- constants (match setup_inputs) ----------------
#define BB 8        // batch
#define CH 1
#define W 128
#define H 128
#define WG 32
#define NG 1024     // WG*HG
#define DZW 8
#define DZI 32
#define DZB 8
#define CF 32
#define CS 32
#define KK 50       // n_objects_max
#define KFLAT 32768 // CF*CS*CS
#define NDEC 2048   // (CH+1)*CS*CS
#define MROWS 400   // KK*BB
#define PIX 16384   // W*H

typedef __attribute__((ext_vector_type(16))) __bf16 v16bf;
typedef __attribute__((ext_vector_type(8)))  float  v8f;
typedef __attribute__((ext_vector_type(4)))  unsigned int v4u;

union ABu { v4u u[2]; v16bf v; };

static __device__ __forceinline__ float sigmoidf_(float x){ return 1.f/(1.f+expf(-x)); }
static __device__ __forceinline__ float softplusf_(float x){ return fmaxf(x,0.f)+log1pf(expf(-fabsf(x))); }
static __device__ __forceinline__ unsigned short f2bf(float f){
    unsigned u = __float_as_uint(f);
    return (unsigned short)((u + 0x7FFFu + ((u>>16)&1u)) >> 16);
}

// ---------------- K0: background (zbg, kl_zbg, bg_lr GEMM, bilinear upsample + sigmoid) ----
__global__ void bg_kernel(const float* zbg_mu, const float* zbg_std, const float* eps_zbg,
                          const float* w_bg, const float* b_bg,
                          float* big_bg, float* kl_zbg){
    int b = blockIdx.x, tid = threadIdx.x;
    __shared__ float lz[DZB];
    __shared__ float lbg[64];
    if (tid < DZB){
        float mu = zbg_mu[b*DZB+tid], sd = zbg_std[b*DZB+tid];
        lz[tid] = mu + sd*eps_zbg[b*DZB+tid];
        kl_zbg[b*DZB+tid] = 0.5f*(mu*mu + sd*sd - 1.f) - logf(sd + 1e-8f);
    }
    __syncthreads();
    if (tid < 64){
        float acc = b_bg[tid];
        #pragma unroll
        for (int d=0; d<DZB; d++) acc += lz[d]*w_bg[tid*DZB+d];
        lbg[tid] = acc;
    }
    __syncthreads();
    for (int p = tid; p < PIX; p += blockDim.x){
        int X = p >> 7, Y = p & 127;
        float u = (X + 0.5f)*0.0625f - 0.5f;   // 8/128 scale, half-pixel centers
        float v = (Y + 0.5f)*0.0625f - 0.5f;
        float x0f = floorf(u), y0f = floorf(v);
        float wx = u - x0f, wy = v - y0f;
        int x0 = min(max((int)x0f,0),7), x1 = min(max((int)x0f+1,0),7);
        int y0 = min(max((int)y0f,0),7), y1 = min(max((int)y0f+1,0),7);
        float r = lbg[x0*8+y0]*(1-wx)*(1-wy) + lbg[x0*8+y1]*(1-wx)*wy
                + lbg[x1*8+y0]*wx*(1-wy)     + lbg[x1*8+y1]*wx*wy;
        big_bg[b*PIX + p] = sigmoidf_(r);
    }
}

// ---------------- K1: where maps, boxes, q/c/score, kl_zwhere --------------------------
__global__ void where_kernel(const float* zw_mu, const float* zw_std, const float* eps_zw,
                             const float* w_where, const float* b_where,
                             const float* logit_mu, const float* u_noise,
                             float* bb, float* x1a, float* x2a, float* y1a, float* y2a,
                             float* areaa, float* scorea, float* qa, float* ca, float* klzw){
    int g = blockIdx.x*blockDim.x + threadIdx.x;
    if (g >= BB*NG) return;
    int b = g >> 10, n = g & 1023;
    float zw[DZW];
    #pragma unroll
    for (int c=0; c<DZW; c++){
        int ix = b*DZW*NG + c*NG + n;
        float mu = zw_mu[ix], sd = zw_std[ix];
        zw[c] = mu + sd*eps_zw[ix];
        klzw[(b*NG+n)*DZW + c] = 0.5f*(mu*mu + sd*sd - 1.f) - logf(sd + 1e-8f);
    }
    float t[4];
    #pragma unroll
    for (int r=0; r<4; r++){
        float acc = b_where[r];
        #pragma unroll
        for (int c=0; c<DZW; c++) acc += w_where[r*DZW+c]*zw[c];
        t[r] = sigmoidf_(acc);
    }
    int i = n >> 5, j = n & 31;
    float bx = (i + t[0])*4.f;                 // W/Wg = 4
    float by = (j + t[1])*4.f;
    float bw = 8.f + t[2]*40.f;                // size_min + t*(size_max-size_min)
    float bh = 8.f + t[3]*40.f;
    int gi = b*NG + n;
    bb[gi*4+0]=bx; bb[gi*4+1]=by; bb[gi*4+2]=bw; bb[gi*4+3]=bh;
    x1a[gi] = bx - 0.5f*bw;  x2a[gi] = bx + 0.5f*bw;
    y1a[gi] = by - 0.5f*bh;  y2a[gi] = by + 0.5f*bh;
    areaa[gi] = bw*bh;
    float q = sigmoidf_(logit_mu[b*NG + n]);
    float cc = (u_noise[n*BB + b] < q) ? 1.f : 0.f;
    qa[gi]=q; ca[gi]=cc; scorea[gi] = q + cc;
}

// ---------------- K2: greedy NMS scan + kl_logit reduction -----------------------------
__global__ void nms_kernel(const float* x1a, const float* x2a, const float* y1a, const float* y2a,
                           const float* areaa, const float* scorea, const float* qa,
                           int* idxs, float* kl_logit){
    int b = blockIdx.x, n = threadIdx.x;
    __shared__ float sx1[NG], sx2[NG], sy1[NG], sy2[NG], sar[NG];
    __shared__ float swv[32];
    __shared__ int   swi[32];
    __shared__ float bvals[5];
    int gi = b*NG + n;
    float mx1=x1a[gi], mx2=x2a[gi], my1=y1a[gi], my2=y2a[gi], mar=areaa[gi];
    sx1[n]=mx1; sx2[n]=mx2; sy1[n]=my1; sy2[n]=my2; sar[n]=mar;
    float myscore = scorea[gi];
    // kl_logit reduction
    float q = qa[gi];
    float kl = q*logf(q*2.f + 1e-8f) + (1.f-q)*logf((1.f-q)*2.f + 1e-8f);
    for (int off=16; off>0; off>>=1) kl += __shfl_down(kl, off, 32);
    if ((n&31)==0) swv[n>>5] = kl;
    __syncthreads();
    if (n < 32){
        float v = swv[n];
        for (int off=16; off>0; off>>=1) v += __shfl_down(v, off, 32);
        if (n==0) kl_logit[b] = v;
    }
    __syncthreads();
    float possible = 1.f;
    for (int k=0; k<KK; k++){
        float v = (possible > 0.5f) ? myscore : -__builtin_inff();
        int idx = n;
        for (int off=16; off>0; off>>=1){
            float ov = __shfl_down(v, off, 32);
            int   oi = __shfl_down(idx, off, 32);
            if (ov > v || (ov == v && oi < idx)){ v = ov; idx = oi; }
        }
        if ((n&31)==0){ swv[n>>5]=v; swi[n>>5]=idx; }
        __syncthreads();                                   // (A)
        if (n < 32){
            float v2 = swv[n]; int i2 = swi[n];
            for (int off=16; off>0; off>>=1){
                float ov = __shfl_down(v2, off, 32);
                int   oi = __shfl_down(i2, off, 32);
                if (ov > v2 || (ov == v2 && oi < i2)){ v2 = ov; i2 = oi; }
            }
            if (n==0){
                idxs[k*BB + b] = i2;
                bvals[0]=sx1[i2]; bvals[1]=sx2[i2]; bvals[2]=sy1[i2];
                bvals[3]=sy2[i2]; bvals[4]=sar[i2];
            }
        }
        __syncthreads();                                   // (B)
        float iw = fmaxf(fminf(bvals[1], mx2) - fmaxf(bvals[0], mx1), 0.f);
        float ih = fmaxf(fminf(bvals[3], my2) - fmaxf(bvals[2], my1), 0.f);
        float inter = iw*ih;
        float iou = inter / (bvals[4] + mar - inter + 1e-6f);
        if (iou >= 0.3f) possible = 0.f;
    }
}

// ---------------- K3: gather the K chosen boxes ----------------------------------------
__global__ void gather_kernel(const int* idxs, const float* qa, const float* ca,
                              const float* bb, const float* klzw,
                              float* q_few, float* c_few, float* bb_few, float* klzw_few){
    int t = blockIdx.x*blockDim.x + threadIdx.x;
    if (t >= MROWS) return;
    int b = t & 7;
    int idx = idxs[t];
    int gi = b*NG + idx;
    q_few[t] = qa[gi];
    c_few[t] = ca[gi];
    #pragma unroll
    for (int r=0;r<4;r++) bb_few[t*4+r] = bb[gi*4+r];
    #pragma unroll
    for (int c=0;c<DZW;c++) klzw_few[t*DZW+c] = klzw[gi*DZW+c];
}

// ---------------- K4: f32 -> bf16 weight conversion ------------------------------------
__global__ void cvt_bf16_kernel(const float* src, unsigned short* dst, int n){
    int i = blockIdx.x*blockDim.x + threadIdx.x;
    if (i < n) dst[i] = f2bf(src[i]);
}

// ---------------- K5: bilinear crop of features -> bf16 activations --------------------
__global__ void crop_kernel(const float* features, const float* bb_few, unsigned short* croppedBf){
    int t = blockIdx.x*blockDim.x + threadIdx.x;
    if (t >= MROWS*CS*CS) return;
    int cy = t & 31, cx = (t >> 5) & 31, mb = t >> 10;   // mb = k*8+b
    int b = mb & 7;
    float bx = bb_few[mb*4+0], by = bb_few[mb*4+1], bw = bb_few[mb*4+2], bh = bb_few[mb*4+3];
    float xs = bx - 0.5f*bw + (cx + 0.5f)*bw*(1.f/CS) - 0.5f;
    float ys = by - 0.5f*bh + (cy + 0.5f)*bh*(1.f/CS) - 0.5f;
    float x0f = fminf(fmaxf(floorf(xs),0.f),(float)(W-2));
    float y0f = fminf(fmaxf(floorf(ys),0.f),(float)(H-2));
    int x0=(int)x0f, y0=(int)y0f;
    float wx = fminf(fmaxf(xs-x0f,0.f),1.f);
    float wy = fminf(fmaxf(ys-y0f,0.f),1.f);
    float w00=(1-wx)*(1-wy), w01=(1-wx)*wy, w10=wx*(1-wy), w11=wx*wy;
    const float* fb = features + (size_t)b*CF*PIX;
    size_t obase = (size_t)mb*KFLAT + cx*CS + cy;
    #pragma unroll 4
    for (int c=0;c<CF;c++){
        const float* p = fb + (size_t)c*PIX + x0*H + y0;
        float val = p[0]*w00 + p[1]*w01 + p[H]*w10 + p[H+1]*w11;
        croppedBf[obase + (size_t)c*CS*CS] = f2bf(val);
    }
}

// ---------------- K6: encoder GEMM (bf16 WMMA, f32 accumulate) -------------------------
// grid (25, 4): blockIdx.y = {mu-n0, mu-n1, std-n0, std-n1}; one wave per 16x16 D tile.
__global__ void enc_wmma_kernel(const unsigned short* croppedBf,
                                const unsigned short* wmu_bf, const unsigned short* wstd_bf,
                                float* muArr, float* srawArr){
    int mtile = blockIdx.x, sel = blockIdx.y, lane = threadIdx.x;
    int wsel = sel >> 1, nt = sel & 1;
    int g = lane >> 4, ln = lane & 15;
    const unsigned short* Arow = croppedBf + (size_t)(mtile*16 + ln) * KFLAT;
    const unsigned short* Brow = (wsel ? wstd_bf : wmu_bf) + (size_t)(nt*16 + ln) * KFLAT;
    v8f acc = {};
    for (int k0 = 0; k0 < KFLAT; k0 += 32){
        ABu a, bm;
        // A 16x32 bf16: lanes 0-15 hold K=[0..7]+[16..23]; lanes 16-31 K=[8..15]+[24..31]
        a.u[0] = *(const v4u*)(Arow + k0 + 8*g);
        a.u[1] = *(const v4u*)(Arow + k0 + 16 + 8*g);
        // B 32x16 bf16: lanes 0-15 hold K=[0..7]+[16..23]; lanes 16-31 K=[8..15]+[24..31]
        bm.u[0] = *(const v4u*)(Brow + k0 + 8*g);
        bm.u[1] = *(const v4u*)(Brow + k0 + 16 + 8*g);
        acc = __builtin_amdgcn_wmma_f32_16x16x32_bf16(false, a.v, false, bm.v,
                                                      (short)0, acc, false, false);
    }
    float* dst = wsel ? srawArr : muArr;
    #pragma unroll
    for (int v=0; v<8; v++){
        int m = mtile*16 + v + 8*g;            // D: VGPR v, lane group g -> row v+8g
        dst[m*DZI + nt*16 + ln] = acc[v];
    }
}

// ---------------- K7: z_instance, kl_zinstance, bf16 z ---------------------------------
__global__ void zinst_kernel(const float* muArr, const float* srawArr, const float* eps_zi,
                             unsigned short* zbf, float* kl_zi){
    int t = blockIdx.x*blockDim.x + threadIdx.x;
    if (t >= MROWS*DZI) return;
    float mu = muArr[t];
    float std = softplusf_(srawArr[t]) + 0.001f;
    float z = mu + std*eps_zi[t];
    kl_zi[t] = 0.5f*(mu*mu + std*std - 1.f) - logf(std + 1e-8f);
    zbf[t] = f2bf(z);
}

// ---------------- K8: decoder GEMM (single-step bf16 WMMA) + bias + activation ---------
__global__ void dec_wmma_kernel(const unsigned short* zbf, const unsigned short* wdec_bf,
                                const float* b_dec, float* smallArr){
    int mtile = blockIdx.x, ntile = blockIdx.y, lane = threadIdx.x;
    int g = lane >> 4, ln = lane & 15;
    const unsigned short* Arow = zbf + (size_t)(mtile*16 + ln) * DZI;
    int n = ntile*16 + ln;
    const unsigned short* Brow = wdec_bf + (size_t)n * DZI;
    ABu a, bm;
    a.u[0] = *(const v4u*)(Arow + 8*g);
    a.u[1] = *(const v4u*)(Arow + 16 + 8*g);
    bm.u[0] = *(const v4u*)(Brow + 8*g);
    bm.u[1] = *(const v4u*)(Brow + 16 + 8*g);
    v8f acc = {};
    acc = __builtin_amdgcn_wmma_f32_16x16x32_bf16(false, a.v, false, bm.v,
                                                  (short)0, acc, false, false);
    float bias = b_dec[n];
    int cidx = n >> 10;                        // 0 -> weight ch (softplus), 1 -> img ch (sigmoid)
    #pragma unroll
    for (int v=0; v<8; v++){
        float x = acc[v] + bias;
        float y = (cidx == 0) ? softplusf_(x) : sigmoidf_(x);
        smallArr[(size_t)(mtile*16 + v + 8*g)*NDEC + n] = y;
    }
}

// ---------------- K9: uncrop (bilinear splat back to full res) -------------------------
__global__ void uncrop_kernel(const float* smallArr, const float* bb_few,
                              float* big_w /*out0, temp weight*/, float* big_img /*out1*/){
    int t = blockIdx.x*blockDim.x + threadIdx.x;
    if (t >= MROWS*PIX) return;
    int p = t & (PIX-1), mb = t >> 14;
    int X = p >> 7, Y = p & 127;
    float bx = bb_few[mb*4+0], by = bb_few[mb*4+1], bw = bb_few[mb*4+2], bh = bb_few[mb*4+3];
    float us = (X + 0.5f - (bx - 0.5f*bw)) * ((float)CS/bw) - 0.5f;
    float vs = (Y + 0.5f - (by - 0.5f*bh)) * ((float)CS/bh) - 0.5f;
    bool valid = (us >= 0.f) && (us <= (float)(CS-1)) && (vs >= 0.f) && (vs <= (float)(CS-1));
    float wgt = 0.f, im = 0.f;
    if (valid){
        float x0f = fminf(fmaxf(floorf(us),0.f),(float)(CS-2));
        float y0f = fminf(fmaxf(floorf(vs),0.f),(float)(CS-2));
        int x0=(int)x0f, y0=(int)y0f;
        float wx = fminf(fmaxf(us-x0f,0.f),1.f);
        float wy = fminf(fmaxf(vs-y0f,0.f),1.f);
        float w00=(1-wx)*(1-wy), w01=(1-wx)*wy, w10=wx*(1-wy), w11=wx*wy;
        const float* s0 = smallArr + (size_t)mb*NDEC;
        int i00 = x0*CS + y0;
        wgt = s0[i00]*w00 + s0[i00+1]*w01 + s0[i00+CS]*w10 + s0[i00+CS+1]*w11;
        const float* s1 = s0 + CS*CS;
        im  = s1[i00]*w00 + s1[i00+1]*w01 + s1[i00+CS]*w10 + s1[i00+CS+1]*w11;
    }
    big_w[t] = wgt;
    big_img[t] = im;
}

// ---------------- K10: mask composition (in-place over out0, writes out2) --------------
__global__ void compose_kernel(float* big_mask_w, float* big_mask_non){
    int t = blockIdx.x*blockDim.x + threadIdx.x;
    if (t >= BB*PIX) return;
    float sum = 0.f;
    for (int k=0;k<KK;k++) sum += big_mask_w[(size_t)k*BB*PIX + t];
    float scale = tanhf(sum) / fmaxf(sum, 1e-6f);
    for (int k=0;k<KK;k++){
        size_t ix = (size_t)k*BB*PIX + t;
        float w = big_mask_w[ix];
        big_mask_w[ix]    = w * scale;
        big_mask_non[ix]  = tanhf(w);
    }
}

// ---------------- host orchestration ---------------------------------------------------
extern "C" void kernel_launch(void* const* d_in, const int* in_sizes, int n_in,
                              void* d_out, int out_size, void* d_ws, size_t ws_size,
                              hipStream_t stream) {
    (void)in_sizes; (void)n_in; (void)out_size; (void)ws_size;
    const float* features   = (const float*)d_in[1];
    const float* logit_mu   = (const float*)d_in[2];
    const float* zwhere_mu  = (const float*)d_in[3];
    const float* zwhere_std = (const float*)d_in[4];
    const float* eps_zwhere = (const float*)d_in[5];
    const float* u_noise    = (const float*)d_in[6];
    const float* zbg_mu     = (const float*)d_in[7];
    const float* zbg_std    = (const float*)d_in[8];
    const float* eps_zbg    = (const float*)d_in[9];
    const float* w_where    = (const float*)d_in[10];
    const float* b_where    = (const float*)d_in[11];
    const float* w_bg       = (const float*)d_in[12];
    const float* b_bg       = (const float*)d_in[13];
    const float* w_enc_mu   = (const float*)d_in[14];
    const float* w_enc_std  = (const float*)d_in[15];
    const float* eps_zi     = (const float*)d_in[16];
    const float* w_dec      = (const float*)d_in[17];
    const float* b_dec      = (const float*)d_in[18];

    // output layout (flat concat in return order)
    float* out = (float*)d_out;
    float* o_big_mask     = out;                         // 50*8*1*128*128
    float* o_big_img      = o_big_mask     + (size_t)MROWS*PIX;
    float* o_big_mask_non = o_big_img      + (size_t)MROWS*PIX;
    float* o_big_bg       = o_big_mask_non + (size_t)MROWS*PIX;
    float* o_q_few        = o_big_bg       + (size_t)BB*PIX;
    float* o_c_few        = o_q_few        + MROWS;
    float* o_bb_few       = o_c_few        + MROWS;
    float* o_klzw_few     = o_bb_few       + MROWS*4;
    float* o_klzi         = o_klzw_few     + MROWS*DZW;
    float* o_klzbg        = o_klzi         + MROWS*DZI;
    float* o_kllogit      = o_klzbg        + BB*DZB;

    // workspace carving (256B aligned slabs)
    char* wp = (char*)d_ws;
    auto alloc = [&](size_t bytes)->void* {
        void* p = (void*)wp; wp += (bytes + 255) & ~(size_t)255; return p;
    };
    float* ws_bb    = (float*)alloc((size_t)BB*NG*4*sizeof(float));
    float* ws_x1    = (float*)alloc((size_t)BB*NG*sizeof(float));
    float* ws_x2    = (float*)alloc((size_t)BB*NG*sizeof(float));
    float* ws_y1    = (float*)alloc((size_t)BB*NG*sizeof(float));
    float* ws_y2    = (float*)alloc((size_t)BB*NG*sizeof(float));
    float* ws_area  = (float*)alloc((size_t)BB*NG*sizeof(float));
    float* ws_score = (float*)alloc((size_t)BB*NG*sizeof(float));
    float* ws_q     = (float*)alloc((size_t)BB*NG*sizeof(float));
    float* ws_c     = (float*)alloc((size_t)BB*NG*sizeof(float));
    float* ws_klzw  = (float*)alloc((size_t)BB*NG*DZW*sizeof(float));
    int*   ws_idxs  = (int*)  alloc((size_t)MROWS*sizeof(int));
    unsigned short* ws_crop = (unsigned short*)alloc((size_t)MROWS*KFLAT*sizeof(unsigned short));
    unsigned short* ws_wmu  = (unsigned short*)alloc((size_t)DZI*KFLAT*sizeof(unsigned short));
    unsigned short* ws_wstd = (unsigned short*)alloc((size_t)DZI*KFLAT*sizeof(unsigned short));
    unsigned short* ws_wdec = (unsigned short*)alloc((size_t)NDEC*DZI*sizeof(unsigned short));
    float* ws_mu    = (float*)alloc((size_t)MROWS*DZI*sizeof(float));
    float* ws_sraw  = (float*)alloc((size_t)MROWS*DZI*sizeof(float));
    unsigned short* ws_zbf = (unsigned short*)alloc((size_t)MROWS*DZI*sizeof(unsigned short));
    float* ws_small = (float*)alloc((size_t)MROWS*NDEC*sizeof(float));

    // pipeline
    bg_kernel<<<BB, 256, 0, stream>>>(zbg_mu, zbg_std, eps_zbg, w_bg, b_bg, o_big_bg, o_klzbg);

    where_kernel<<<(BB*NG)/256, 256, 0, stream>>>(zwhere_mu, zwhere_std, eps_zwhere,
        w_where, b_where, logit_mu, u_noise,
        ws_bb, ws_x1, ws_x2, ws_y1, ws_y2, ws_area, ws_score, ws_q, ws_c, ws_klzw);

    nms_kernel<<<BB, NG, 0, stream>>>(ws_x1, ws_x2, ws_y1, ws_y2, ws_area, ws_score, ws_q,
                                      ws_idxs, o_kllogit);

    gather_kernel<<<2, 256, 0, stream>>>(ws_idxs, ws_q, ws_c, ws_bb, ws_klzw,
                                         o_q_few, o_c_few, o_bb_few, o_klzw_few);

    cvt_bf16_kernel<<<(DZI*KFLAT)/256, 256, 0, stream>>>(w_enc_mu,  ws_wmu,  DZI*KFLAT);
    cvt_bf16_kernel<<<(DZI*KFLAT)/256, 256, 0, stream>>>(w_enc_std, ws_wstd, DZI*KFLAT);
    cvt_bf16_kernel<<<(NDEC*DZI)/256, 256, 0, stream>>>(w_dec, ws_wdec, NDEC*DZI);

    crop_kernel<<<(MROWS*CS*CS)/256, 256, 0, stream>>>(features, o_bb_few, ws_crop);

    enc_wmma_kernel<<<dim3(MROWS/16, 4), 32, 0, stream>>>(ws_crop, ws_wmu, ws_wstd,
                                                          ws_mu, ws_sraw);

    zinst_kernel<<<(MROWS*DZI)/256, 256, 0, stream>>>(ws_mu, ws_sraw, eps_zi, ws_zbf, o_klzi);

    dec_wmma_kernel<<<dim3(MROWS/16, NDEC/16), 32, 0, stream>>>(ws_zbf, ws_wdec, b_dec, ws_small);

    uncrop_kernel<<<(MROWS*PIX)/256, 256, 0, stream>>>(ws_small, o_bb_few,
                                                       o_big_mask, o_big_img);

    compose_kernel<<<(BB*PIX)/256, 256, 0, stream>>>(o_big_mask, o_big_mask_non);
}
